// GatedDeltaNetEncoder_61710090109410
// MI455X (gfx1250) — compile-verified, matrix operand order checked by reading, not run on hardware
//
#include <hip/hip_runtime.h>
#include <hip/hip_bf16.h>

// ---------------------------------------------------------------------------
// GatedDeltaNet encoder layer for gfx1250 (MI455X), wave32 + WMMA.
// GEMMs: v_wmma_f32_16x16x32_f16, 2x2 tiles per wave; B tile staged in LDS
// via global_load_async_to_lds_b128 (double-buffered, ASYNCcnt-tracked).
// Weights pre-transposed to [N][K] f16 so all operand loads are 128-bit.
// ---------------------------------------------------------------------------

typedef __attribute__((ext_vector_type(16))) _Float16 v16h;
typedef __attribute__((ext_vector_type(8)))  _Float16 v8h;
typedef __attribute__((ext_vector_type(8)))  float    v8f;

#define BB 4
#define TT 1024
#define DD 1024
#define HH 16
#define DKK 64
#define DVV 64
#define KC 4
#define II 2816
#define MM (BB * TT)          // 4096 tokens
#define EPSF 1e-6f
#define QSCALE 0.125f         // DK^-0.5

#define BROW 40               // LDS B-tile row stride in halves (80B, bank-spread)

// ---------------------------------------------------------------------------
// Tiled transpose + f32->f16: in [Kd][N] row-major  ->  out [N][Kd] f16.
// ---------------------------------------------------------------------------
__global__ void cvt_transpose_tiled(const float* __restrict__ in, _Float16* __restrict__ out,
                                    int Kd, int N) {
    __shared__ float tile[32][33];
    int tk = blockIdx.y * 32, tn = blockIdx.x * 32;
#pragma unroll
    for (int j = 0; j < 4; ++j) {
        int k = threadIdx.y + j * 8;
        tile[k][threadIdx.x] = in[(size_t)(tk + k) * N + tn + threadIdx.x];
    }
    __syncthreads();
#pragma unroll
    for (int j = 0; j < 4; ++j) {
        int n = threadIdx.y + j * 8;
        out[(size_t)(tn + n) * Kd + tk + threadIdx.x] = (_Float16)tile[threadIdx.x][n];
    }
}

// Naive transpose-convert for small weights (Wb/Wa, N=16).
__global__ void cvt_transpose_naive(const float* __restrict__ in, _Float16* __restrict__ out,
                                    int Kd, int N) {
    int i = blockIdx.x * 256 + threadIdx.x;
    if (i >= Kd * N) return;
    int k = i / N, n = i % N;
    out[(size_t)n * Kd + k] = (_Float16)in[i];
}

// ---------------------------------------------------------------------------
// Fused (optional residual) add + RMSNorm over Dim=1024.
// ---------------------------------------------------------------------------
__global__ void add_rmsnorm(const float* __restrict__ x, const float* __restrict__ add,
                            const float* __restrict__ w, float* __restrict__ sum_out,
                            _Float16* __restrict__ h16) {
    int row = blockIdx.x;
    int tid = threadIdx.x;
    const float* xr = x + (size_t)row * DD;
    float vals[4];
    float ss = 0.f;
#pragma unroll
    for (int j = 0; j < 4; ++j) {
        int c = tid + j * 256;
        float vv = xr[c];
        if (add) vv += add[(size_t)row * DD + c];
        vals[j] = vv;
        ss += vv * vv;
    }
    __shared__ float red[256];
    red[tid] = ss;
    __syncthreads();
    for (int s = 128; s > 0; s >>= 1) {
        if (tid < s) red[tid] += red[tid + s];
        __syncthreads();
    }
    float rs = rsqrtf(red[0] / (float)DD + EPSF);
#pragma unroll
    for (int j = 0; j < 4; ++j) {
        int c = tid + j * 256;
        if (sum_out) sum_out[(size_t)row * DD + c] = vals[j];
        h16[(size_t)row * DD + c] = (_Float16)(vals[j] * rs * w[c]);
    }
}

// ---------------------------------------------------------------------------
// WMMA GEMM: C[M,N] = A[M,Kd] (f16 row-major) x B (given as Bt[N][Kd] f16).
// Block = 4 waves stacked along M; all waves share one 32-col B tile which is
// double-buffered through LDS with global_load_async_to_lds_b128. Each wave
// computes a 32x32 patch (2x2 WMMA tiles). Branch-free K-loop; Bt must be
// padded to >=32 rows (garbage columns computed but never stored).
// outF16: 0 -> f32 C, 1 -> f16 C. Wave-level control flow is lane-uniform.
// ---------------------------------------------------------------------------
__global__ void wmma_gemm_f16t(const _Float16* __restrict__ A, const _Float16* __restrict__ Bt,
                               void* __restrict__ Cout, int M, int N, int Kd, int outF16) {
    const int lane = threadIdx.x;                      // 0..31
    const int wave = threadIdx.y;                      // 0..3
    const int tn = blockIdx.x * 32;
    const int tm = (blockIdx.y * 4 + wave) * 32;
    const bool haveN2 = (tn + 16) < N;                 // wave-uniform store guard

    __shared__ _Float16 bsm[2][32 * BROW];             // double-buffered B tile

    // async staging: 128 threads x one b128 transfer = 32 rows x 64B
    const int tFlat = wave * 32 + lane;
    const int sr = tFlat >> 2;                         // 0..31 (row of B tile)
    const int sc = tFlat & 3;                          // 16B chunk within row
    const _Float16* bsrc = Bt + (size_t)(tn + sr) * Kd + sc * 8;

    const int row  = lane & 15;
    const int koff = (lane >> 4) << 4;                 // 0 or 16 (K split)
    const _Float16* a0p = A + (size_t)(tm + row) * Kd + koff;
    const _Float16* a1p = A + (size_t)(tm + 16 + row) * Kd + koff;
    const _Float16* lb0 = &bsm[0][(size_t)row * BROW + koff];
    const _Float16* lb1 = &bsm[0][(size_t)(16 + row) * BROW + koff];

    v8f acc00 = {}, acc01 = {}, acc10 = {}, acc11 = {};

    // prologue: stage K-step 0 into buffer 0
    {
        unsigned ldsOff = (unsigned)(uintptr_t)&bsm[0][sr * BROW + sc * 8];
        asm volatile("global_load_async_to_lds_b128 %0, %1, off"
                     :: "v"(ldsOff), "v"(bsrc) : "memory");
    }
    asm volatile("s_wait_asynccnt 0x0" ::: "memory");
    __syncthreads();

    int cur = 0;
    for (int kk = 0; kk < Kd; kk += 32) {
        // stage next K-step into the other buffer (uniform branch)
        if (kk + 32 < Kd) {
            unsigned ldsOff = (unsigned)(uintptr_t)&bsm[cur ^ 1][sr * BROW + sc * 8];
            const _Float16* g = bsrc + kk + 32;
            asm volatile("global_load_async_to_lds_b128 %0, %1, off"
                         :: "v"(ldsOff), "v"(g) : "memory");
        }

        union { v16h v; v8h h[2]; } a0, a1, b0, b1;
        a0.h[0] = *(const v8h*)(a0p + kk);
        a0.h[1] = *(const v8h*)(a0p + kk + 8);
        a1.h[0] = *(const v8h*)(a1p + kk);
        a1.h[1] = *(const v8h*)(a1p + kk + 8);
        __builtin_prefetch(a0p + kk + 32, 0, 1);
        __builtin_prefetch(a1p + kk + 32, 0, 1);

        const _Float16* l0 = lb0 + (size_t)cur * (32 * BROW);
        const _Float16* l1 = lb1 + (size_t)cur * (32 * BROW);
        b0.h[0] = *(const v8h*)(l0);
        b0.h[1] = *(const v8h*)(l0 + 8);
        b1.h[0] = *(const v8h*)(l1);
        b1.h[1] = *(const v8h*)(l1 + 8);

        acc00 = __builtin_amdgcn_wmma_f32_16x16x32_f16(false, a0.v, false, b0.v,
                                                       (short)0, acc00, false, false);
        acc10 = __builtin_amdgcn_wmma_f32_16x16x32_f16(false, a1.v, false, b0.v,
                                                       (short)0, acc10, false, false);
        acc01 = __builtin_amdgcn_wmma_f32_16x16x32_f16(false, a0.v, false, b1.v,
                                                       (short)0, acc01, false, false);
        acc11 = __builtin_amdgcn_wmma_f32_16x16x32_f16(false, a1.v, false, b1.v,
                                                       (short)0, acc11, false, false);

        asm volatile("s_wait_asynccnt 0x0" ::: "memory");
        __syncthreads();
        cur ^= 1;
    }

    // C/D layout: lanes 0-15 -> M = r, lanes 16-31 -> M = 8 + r  (r = VGPR idx)
    const int mo  = (lane >> 4) << 3;
    const int col = lane & 15;
    if (outF16) {
        _Float16* C = (_Float16*)Cout;
#pragma unroll
        for (int r = 0; r < 8; ++r) {
            C[(size_t)(tm + mo + r) * N + tn + col]      = (_Float16)acc00[r];
            C[(size_t)(tm + 16 + mo + r) * N + tn + col] = (_Float16)acc10[r];
        }
        if (haveN2) {
#pragma unroll
            for (int r = 0; r < 8; ++r) {
                C[(size_t)(tm + mo + r) * N + tn + 16 + col]      = (_Float16)acc01[r];
                C[(size_t)(tm + 16 + mo + r) * N + tn + 16 + col] = (_Float16)acc11[r];
            }
        }
    } else {
        float* C = (float*)Cout;
#pragma unroll
        for (int r = 0; r < 8; ++r) {
            C[(size_t)(tm + mo + r) * N + tn + col]      = acc00[r];
            C[(size_t)(tm + 16 + mo + r) * N + tn + col] = acc10[r];
        }
        if (haveN2) {
#pragma unroll
            for (int r = 0; r < 8; ++r) {
                C[(size_t)(tm + mo + r) * N + tn + 16 + col]      = acc01[r];
                C[(size_t)(tm + 16 + mo + r) * N + tn + 16 + col] = acc11[r];
            }
        }
    }
}

// ---------------------------------------------------------------------------
// Causal depthwise conv1d (K=4) + SiLU. pre: f16 [B,T,C], w: f32 [C,K].
// ---------------------------------------------------------------------------
__global__ void conv_silu(const _Float16* __restrict__ pre, const float* __restrict__ w,
                          float* __restrict__ out) {
    int bt = blockIdx.x;
    int b = bt / TT, t = bt % TT;
#pragma unroll
    for (int j = 0; j < 4; ++j) {
        int c = threadIdx.x + j * 256;
        float acc = 0.f;
#pragma unroll
        for (int i = 0; i < KC; ++i) {
            int tt = t - (KC - 1) + i;
            if (tt >= 0)
                acc += (float)pre[((size_t)(b * TT + tt)) * DD + c] * w[c * KC + i];
        }
        out[(size_t)bt * DD + c] = acc / (1.f + __expf(-acc));   // SiLU
    }
}

// ---------------------------------------------------------------------------
// Per-head l2norm of q (x SCALE) and k. grid = M*H, block = 64.
// ---------------------------------------------------------------------------
__global__ void l2norm_qk(float* __restrict__ q, float* __restrict__ k) {
    size_t row = blockIdx.x;                 // (b*T+t)*H + h
    int lane = threadIdx.x;
    size_t idx = (row / HH) * DD + (row % HH) * DKK + lane;
    float qv = q[idx], kv = k[idx];
    __shared__ float rq[64], rk[64];
    rq[lane] = qv * qv;
    rk[lane] = kv * kv;
    __syncthreads();
    for (int s = 32; s > 0; s >>= 1) {
        if (lane < s) { rq[lane] += rq[lane + s]; rk[lane] += rk[lane + s]; }
        __syncthreads();
    }
    q[idx] = qv * rsqrtf(rq[0] + EPSF) * QSCALE;
    k[idx] = kv * rsqrtf(rk[0] + EPSF);
}

// ---------------------------------------------------------------------------
// beta = 2*sigmoid(pb); g = -exp(A_log) * softplus(pa + dt_bias)
// ---------------------------------------------------------------------------
__global__ void beta_gate(const float* __restrict__ pb, const float* __restrict__ pa,
                          const float* __restrict__ dt_bias, const float* __restrict__ A_log,
                          float* __restrict__ beta, float* __restrict__ g) {
    int i = blockIdx.x * 256 + threadIdx.x;
    if (i >= MM * HH) return;
    int h = i & (HH - 1);
    beta[i] = 2.f / (1.f + __expf(-pb[i]));
    float z = pa[i] + dt_bias[h];
    float sp = (z > 20.f) ? z : log1pf(__expf(z));
    g[i] = -__expf(A_log[h]) * sp;
}

// ---------------------------------------------------------------------------
// Sequential gated delta-rule scan. One block per (b,h) chain (64 blocks),
// 256 threads; thread (kg, v) owns S[kg*16 .. kg*16+15][v] in registers.
// ---------------------------------------------------------------------------
__global__ void delta_scan(const float* __restrict__ q, const float* __restrict__ k,
                           const float* __restrict__ v, const float* __restrict__ beta,
                           const float* __restrict__ g, float* __restrict__ o,
                           float* __restrict__ state_out) {
    int b = blockIdx.x / HH, h = blockIdx.x % HH;
    int tid = threadIdx.x;
    int vcol = tid & 63;
    int kg = tid >> 6;                        // 0..3
    float S[16];
#pragma unroll
    for (int i = 0; i < 16; ++i) S[i] = 0.f;

    __shared__ float sq[64], sk[64], sv[64], sdelta[64], red[256], sbg[2];

    for (int t = 0; t < TT; ++t) {
        size_t rowbase = ((size_t)(b * TT + t)) * DD + h * DKK;
        if (tid < 64)            sq[tid]       = q[rowbase + tid];
        else if (tid < 128)      sk[tid - 64]  = k[rowbase + (tid - 64)];
        else if (tid < 192)      sv[tid - 128] = v[rowbase + (tid - 128)];
        else if (tid == 192) {
            size_t bg = (size_t)(b * TT + t) * HH + h;
            sbg[0] = beta[bg];
            sbg[1] = __expf(g[bg]);
        }
        __syncthreads();

        float dec = sbg[1], bt = sbg[0];
        float partial = 0.f;
#pragma unroll
        for (int i = 0; i < 16; ++i) {
            S[i] *= dec;
            partial += sk[kg * 16 + i] * S[i];
        }
        red[tid] = partial;
        __syncthreads();
        if (tid < 64) {
            float kv = red[tid] + red[64 + tid] + red[128 + tid] + red[192 + tid];
            sdelta[tid] = (sv[tid] - kv) * bt;
        }
        __syncthreads();

        float dl = sdelta[vcol];
        float op = 0.f;
#pragma unroll
        for (int i = 0; i < 16; ++i) {
            S[i] += sk[kg * 16 + i] * dl;
            op += sq[kg * 16 + i] * S[i];
        }
        red[tid] = op;
        __syncthreads();
        if (tid < 64)
            o[rowbase + tid] = red[tid] + red[64 + tid] + red[128 + tid] + red[192 + tid];
        __syncthreads();
    }

#pragma unroll
    for (int i = 0; i < 16; ++i)
        state_out[(((size_t)b * HH + h) * DKK + kg * 16 + i) * DVV + vcol] = S[i];
}

// ---------------------------------------------------------------------------
// Per-head output RMSNorm over DV=64 -> f16. grid = M*H, block = 64.
// ---------------------------------------------------------------------------
__global__ void o_rmsnorm(const float* __restrict__ o, const float* __restrict__ w,
                          _Float16* __restrict__ out16) {
    size_t row = blockIdx.x;
    int lane = threadIdx.x;
    size_t idx = (row / HH) * DD + (row % HH) * DVV + lane;
    float ov = o[idx];
    __shared__ float red[64];
    red[lane] = ov * ov;
    __syncthreads();
    for (int s = 32; s > 0; s >>= 1) {
        if (lane < s) red[lane] += red[lane + s];
        __syncthreads();
    }
    out16[idx] = (_Float16)(ov * rsqrtf(red[0] / (float)DVV + EPSF) * w[lane]);
}

// ---------------------------------------------------------------------------
// SwiGLU: act = silu(gy[:, :I]) * gy[:, I:2I]   (f16 in/out)
// ---------------------------------------------------------------------------
__global__ void swiglu(const _Float16* __restrict__ gy, _Float16* __restrict__ act) {
    long i = (long)blockIdx.x * 256 + threadIdx.x;
    if (i >= (long)MM * II) return;
    long r = i / II, c = i % II;
    float gv = (float)gy[r * (2 * II) + c];
    float yv = (float)gy[r * (2 * II) + II + c];
    act[i] = (_Float16)((gv / (1.f + __expf(-gv))) * yv);
}

// ---------------------------------------------------------------------------
// out = a + b
// ---------------------------------------------------------------------------
__global__ void add2(const float* __restrict__ a, const float* __restrict__ b,
                     float* __restrict__ out, long n) {
    long i = (long)blockIdx.x * 256 + threadIdx.x;
    if (i < n) out[i] = a[i] + b[i];
}

// ---------------------------------------------------------------------------
// Host launcher
// ---------------------------------------------------------------------------
extern "C" void kernel_launch(void* const* d_in, const int* in_sizes, int n_in,
                              void* d_out, int out_size, void* d_ws, size_t ws_size,
                              hipStream_t stream) {
    const float* x        = (const float*)d_in[0];
    const float* attn_nw  = (const float*)d_in[1];
    const float* Wq       = (const float*)d_in[2];
    const float* Wk       = (const float*)d_in[3];
    const float* Wv       = (const float*)d_in[4];
    const float* cqw      = (const float*)d_in[5];
    const float* ckw      = (const float*)d_in[6];
    const float* cvw      = (const float*)d_in[7];
    const float* Wb       = (const float*)d_in[8];
    const float* Wa       = (const float*)d_in[9];
    const float* dt_bias  = (const float*)d_in[10];
    const float* A_log    = (const float*)d_in[11];
    const float* o_nw     = (const float*)d_in[12];
    const float* Wo       = (const float*)d_in[13];
    const float* mlp_nw   = (const float*)d_in[14];
    const float* Wg       = (const float*)d_in[15];
    const float* Wd       = (const float*)d_in[16];

    float* out = (float*)d_out;                         // [M*D] then [B,H,DK,DV]
    float* state_out = out + (size_t)MM * DD;

    // ---- workspace carve-up (256B aligned) ----
    char* ws = (char*)d_ws;
    size_t off = 0;
    auto alloc = [&](size_t bytes) -> char* {
        char* p = ws + off;
        off += (bytes + 255) & ~(size_t)255;
        return p;
    };
    // transposed f16 weights: [N][K]  (Wb/Wa padded to 32 rows for the GEMM)
    _Float16* Wq16 = (_Float16*)alloc((size_t)DD * DD * 2);
    _Float16* Wk16 = (_Float16*)alloc((size_t)DD * DD * 2);
    _Float16* Wv16 = (_Float16*)alloc((size_t)DD * DD * 2);
    _Float16* Wb16 = (_Float16*)alloc((size_t)32 * DD * 2);
    _Float16* Wa16 = (_Float16*)alloc((size_t)32 * DD * 2);
    _Float16* Wo16 = (_Float16*)alloc((size_t)DD * DD * 2);
    _Float16* Wg16 = (_Float16*)alloc((size_t)DD * 2 * II * 2);
    _Float16* Wd16 = (_Float16*)alloc((size_t)II * DD * 2);
    _Float16* h16  = (_Float16*)alloc((size_t)MM * DD * 2);
    _Float16* qpre = (_Float16*)alloc((size_t)MM * DD * 2);
    _Float16* kpre = (_Float16*)alloc((size_t)MM * DD * 2);
    _Float16* vpre = (_Float16*)alloc((size_t)MM * DD * 2);
    float*    qf   = (float*)alloc((size_t)MM * DD * 4);
    float*    kf   = (float*)alloc((size_t)MM * DD * 4);
    float*    vf   = (float*)alloc((size_t)MM * DD * 4);
    float*    bpre = (float*)alloc((size_t)MM * HH * 4);
    float*    apre = (float*)alloc((size_t)MM * HH * 4);
    float*    beta = (float*)alloc((size_t)MM * HH * 4);
    float*    gbuf = (float*)alloc((size_t)MM * HH * 4);
    float*    obuf = (float*)alloc((size_t)MM * DD * 4);
    _Float16* on16 = (_Float16*)alloc((size_t)MM * DD * 2);
    float*    attn = (float*)alloc((size_t)MM * DD * 4);
    float*    h2   = (float*)alloc((size_t)MM * DD * 4);
    _Float16* hm16 = (_Float16*)alloc((size_t)MM * DD * 2);
    _Float16* gy16 = (_Float16*)alloc((size_t)MM * 2 * II * 2);
    _Float16* act16= (_Float16*)alloc((size_t)MM * II * 2);
    float*    mlp  = (float*)alloc((size_t)MM * DD * 4);

    auto cvtT = [&](const float* src, _Float16* dst, int Kd, int N) {
        dim3 blk(32, 8);
        dim3 grd((unsigned)(N / 32), (unsigned)(Kd / 32));
        cvt_transpose_tiled<<<grd, blk, 0, stream>>>(src, dst, Kd, N);
    };
    auto gemm = [&](const _Float16* A, const _Float16* Bt, void* C,
                    int Mr, int Nc, int Kd, int outF16) {
        dim3 blk(32, 4);
        dim3 grd((unsigned)((Nc + 31) / 32), (unsigned)(Mr / 128));
        wmma_gemm_f16t<<<grd, blk, 0, stream>>>(A, Bt, C, Mr, Nc, Kd, outF16);
    };

    // 1) weight conversion + transpose: [K][N] f32 -> [N][K] f16
    cvtT(Wq, Wq16, DD, DD);
    cvtT(Wk, Wk16, DD, DD);
    cvtT(Wv, Wv16, DD, DD);
    cvt_transpose_naive<<<dim3((DD * HH + 255) / 256), dim3(256), 0, stream>>>(Wb, Wb16, DD, HH);
    cvt_transpose_naive<<<dim3((DD * HH + 255) / 256), dim3(256), 0, stream>>>(Wa, Wa16, DD, HH);
    cvtT(Wo, Wo16, DD, DD);
    cvtT(Wg, Wg16, DD, 2 * II);
    cvtT(Wd, Wd16, II, DD);

    // 2) h = rmsnorm(x, attn_norm_w)  (f16 operand)
    add_rmsnorm<<<dim3(MM), dim3(256), 0, stream>>>(x, nullptr, attn_nw, nullptr, h16);

    // 3) q/k/v projections (WMMA)
    gemm(h16, Wq16, qpre, MM, DD, DD, 1);
    gemm(h16, Wk16, kpre, MM, DD, DD, 1);
    gemm(h16, Wv16, vpre, MM, DD, DD, 1);

    // 4) gate projections (N=16 tiles; padded B rows, stores guarded)
    gemm(h16, Wb16, bpre, MM, HH, DD, 0);
    gemm(h16, Wa16, apre, MM, HH, DD, 0);

    // 5) causal depthwise conv + SiLU
    conv_silu<<<dim3(MM), dim3(256), 0, stream>>>(qpre, cqw, qf);
    conv_silu<<<dim3(MM), dim3(256), 0, stream>>>(kpre, ckw, kf);
    conv_silu<<<dim3(MM), dim3(256), 0, stream>>>(vpre, cvw, vf);

    // 6) per-head l2norm (q scaled)
    l2norm_qk<<<dim3(MM * HH), dim3(64), 0, stream>>>(qf, kf);

    // 7) beta / forget gate
    beta_gate<<<dim3((MM * HH + 255) / 256), dim3(256), 0, stream>>>(
        bpre, apre, dt_bias, A_log, beta, gbuf);

    // 8) sequential delta-rule scan (also emits last_state into d_out tail)
    delta_scan<<<dim3(BB * HH), dim3(256), 0, stream>>>(qf, kf, vf, beta, gbuf, obuf, state_out);

    // 9) per-head output RMSNorm -> f16
    o_rmsnorm<<<dim3(MM * HH), dim3(64), 0, stream>>>(obuf, o_nw, on16);

    // 10) output projection
    gemm(on16, Wo16, attn, MM, DD, DD, 0);

    // 11) h2 = x + attn ; hm = rmsnorm(h2, mlp_norm_w) (f16)
    add_rmsnorm<<<dim3(MM), dim3(256), 0, stream>>>(x, attn, mlp_nw, h2, hm16);

    // 12) gated MLP up-projection [M, 2I]
    gemm(hm16, Wg16, gy16, MM, 2 * II, DD, 1);

    // 13) SwiGLU
    swiglu<<<dim3((unsigned)(((long)MM * II + 255) / 256)), dim3(256), 0, stream>>>(gy16, act16);

    // 14) down projection [M, D]
    gemm(act16, Wd16, mlp, MM, DD, II, 0);

    // 15) final residual
    add2<<<dim3((unsigned)(((long)MM * DD + 255) / 256)), dim3(256), 0, stream>>>(
        h2, mlp, out, (long)MM * DD);
}
